// MultiHeadAttention_9277129359623
// MI455X (gfx1250) — matrix operand order repaired and checked by reading, not compiled, not run on hardware
//
#include <hip/hip_runtime.h>

#define BDIM   4
#define SEQ    2048
#define DMODEL 1024
#define NHEAD  16
#define DEPTH  64

typedef __attribute__((ext_vector_type(16))) __bf16 bf16x16;
typedef __attribute__((ext_vector_type(8)))  __bf16 bf16x8;
typedef __attribute__((ext_vector_type(8)))  float  f32x8;

static __device__ __forceinline__ f32x8 wmma_bf16(bf16x16 a, bf16x16 b, f32x8 c) {
  return __builtin_amdgcn_wmma_f32_16x16x32_bf16(false, a, false, b, (short)0, c, false, false);
}

// Async global -> LDS copies (ASYNCcnt path, no VGPR round trip).
static __device__ __forceinline__ void async_b64(uint32_t lds_addr, const void* gaddr) {
  asm volatile("global_load_async_to_lds_b64 %0, %1, off"
               :: "v"(lds_addr), "v"(gaddr) : "memory");
}
static __device__ __forceinline__ void async_b32(uint32_t lds_addr, const void* gaddr) {
  asm volatile("global_load_async_to_lds_b32 %0, %1, off"
               :: "v"(lds_addr), "v"(gaddr) : "memory");
}
static __device__ __forceinline__ void wait_async0() {
  asm volatile("s_wait_asynccnt 0x0" ::: "memory");
}
static __device__ __forceinline__ void wait_async1() {
  asm volatile("s_wait_asynccnt 0x1" ::: "memory");
}

// ---------------------------------------------------------------------------
// Kernel 1: weight fp32 [k][n] -> bf16 transposed [n][k], LDS tile transpose.
// ---------------------------------------------------------------------------
__global__ __launch_bounds__(256) void wt_convert_kernel(const float* __restrict__ W,
                                                         __bf16* __restrict__ dst) {
  __shared__ float tile[32][33];
  const int kbase = blockIdx.x * 32;
  const int nbase = blockIdx.y * 32;
  const int tx = threadIdx.x & 31;
  const int ty = threadIdx.x >> 5;
#pragma unroll
  for (int i = 0; i < 32; i += 8)
    tile[ty + i][tx] = W[(size_t)(kbase + ty + i) * DMODEL + nbase + tx];
  __syncthreads();
#pragma unroll
  for (int i = 0; i < 32; i += 8)
    dst[(size_t)(nbase + ty + i) * DMODEL + kbase + tx] = (__bf16)tile[tx][ty + i];
}

// ---------------------------------------------------------------------------
// Kernel 2: projection GEMM  X(fp32 [B*S,D]) @ Wt(bf16,[n][k]) + bias.
// A tile (16x32 fp32) triple-buffered via async-to-LDS (one batch always in
// flight); B fragments register double-buffered one k-step ahead.
// vmode 0: write head-split [B,H,S,64]; vmode 1: write V transposed [B,H,64,S].
// ---------------------------------------------------------------------------
__global__ __launch_bounds__(256) void proj_kernel(const float* __restrict__ X,
                                                   const __bf16* __restrict__ Wt,
                                                   const float* __restrict__ bias,
                                                   __bf16* __restrict__ out,
                                                   int vmode) {
  __shared__ float Abuf[3][16 * 32];                 // 3 x 2KB
  const int tid     = threadIdx.x;
  const int lane    = tid & 31;
  const int wave    = tid >> 5;
  const int rowbase = blockIdx.x * 16;
  const int colbase = blockIdx.y * 512 + wave * 64;
  const int half    = lane >> 4;
  const int ln      = lane & 15;

  // uniform staging: every thread copies 8B; tile row = tid>>4, k = (tid&15)*2
  const float* gsrc = X + (size_t)(rowbase + (tid >> 4)) * DMODEL + (tid & 15) * 2;
  const uint32_t myl = (uint32_t)(uintptr_t)&Abuf[0][0] + (uint32_t)tid * 8u;

  async_b64(myl, gsrc);                              // batch 0
  async_b64(myl + 2048u, gsrc + 32);                 // batch 1

  const __bf16* wbase[4];
#pragma unroll
  for (int t = 0; t < 4; ++t)
    wbase[t] = Wt + (size_t)(colbase + t * 16 + ln) * DMODEL + half * 16;
  bf16x16 bfrag[2][4];
#pragma unroll
  for (int t = 0; t < 4; ++t) bfrag[0][t] = *(const bf16x16*)wbase[t];

  f32x8 acc[4] = {};
  const int NK = DMODEL / 32;                        // 32
#pragma unroll 2
  for (int it = 0; it < NK; ++it) {
    if (it == NK - 1) wait_async0(); else wait_async1();
    __syncthreads();
    if (it + 2 < NK)                                 // keep one full tile in flight
      async_b64(myl + (uint32_t)((it + 2) % 3) * 2048u, gsrc + (it + 2) * 32);
    if (it + 1 < NK) {                               // B prefetch one step ahead
#pragma unroll
      for (int t = 0; t < 4; ++t)
        bfrag[(it + 1) & 1][t] = *(const bf16x16*)(wbase[t] + (it + 1) * 32);
    }
    // A fragment: lanes<16: K={0..15}; lanes>=16: {16..31}
    const float* ar = &Abuf[it % 3][ln * 32 + half * 16];
    f32x8 x0 = *(const f32x8*)ar;
    f32x8 x1 = *(const f32x8*)(ar + 8);
    bf16x16 a;
#pragma unroll
    for (int i = 0; i < 8; ++i) { a[i] = (__bf16)x0[i]; a[8 + i] = (__bf16)x1[i]; }
#pragma unroll
    for (int t = 0; t < 4; ++t)
      acc[t] = wmma_bf16(a, bfrag[it & 1][t], acc[t]);
  }

  const int mr0   = half * 8;
  const int b_idx = rowbase / SEQ;
  const int sbase = rowbase - b_idx * SEQ;
#pragma unroll
  for (int t = 0; t < 4; ++t) {
    int c = colbase + t * 16 + ln;
    float bvv = bias[c];
    int h = c >> 6, dep = c & 63;
#pragma unroll
    for (int r = 0; r < 8; ++r) {
      float val = acc[t][r] + bvv;
      int s = sbase + mr0 + r;
      size_t off;
      if (vmode == 0)
        off = ((((size_t)b_idx * NHEAD + h) * SEQ + s) << 6) + dep;          // [B,H,S,64]
      else
        off = (((size_t)b_idx * NHEAD + h) * DEPTH + dep) * (size_t)SEQ + s; // [B,H,64,S]
      out[off] = (__bf16)val;
    }
  }
}

// ---------------------------------------------------------------------------
// Kernel 3: flash attention. One wave per 16-row Q tile; KV tiles of 32 keys.
// V loads hoisted above the softmax; base-2 online softmax (v_exp_f32).
// ---------------------------------------------------------------------------
__global__ __launch_bounds__(256) void flash_kernel(const __bf16* __restrict__ Qh,
                                                    const __bf16* __restrict__ Kh,
                                                    const __bf16* __restrict__ Vt,
                                                    __bf16* __restrict__ attn,
                                                    const int* __restrict__ maskp) {
  __shared__ __bf16 plds[8][16 * 32];
  const int lane  = threadIdx.x & 31;
  const int wave  = threadIdx.x >> 5;
  const int g     = blockIdx.x * 8 + wave;
  const int qtile = g & (SEQ / 16 - 1);
  const int bh    = g >> 7;                          // SEQ/16 == 128
  const int qbase = qtile * 16;
  const int half  = lane >> 4;
  const int ln    = lane & 15;
  const int mr0   = half * 8;
  const bool causal = (*maskp != 0);
  const float SCL2 = 0.125f * 1.44269504088896340736f;   // 1/sqrt(64) * log2(e)

  const __bf16* qp = Qh + ((size_t)bh * SEQ + (qbase + ln)) * DEPTH;
  bf16x16 qa0, qa1;
  {
    int ka = half * 16;
    bf16x8 c0 = *(const bf16x8*)(qp + ka);
    bf16x8 c1 = *(const bf16x8*)(qp + ka + 8);
    bf16x8 c2 = *(const bf16x8*)(qp + 32 + ka);
    bf16x8 c3 = *(const bf16x8*)(qp + 32 + ka + 8);
#pragma unroll
    for (int i = 0; i < 8; ++i) {
      qa0[i] = c0[i]; qa0[8 + i] = c1[i];
      qa1[i] = c2[i]; qa1[8 + i] = c3[i];
    }
  }

  float mrow[8], lrow[8];
#pragma unroll
  for (int r = 0; r < 8; ++r) { mrow[r] = -3.0e38f; lrow[r] = 0.0f; }
  f32x8 o[4] = {};

  const int kvend = causal ? (qbase + 16) : SEQ;
  for (int kv = 0; kv < kvend; kv += 32) {
    // ---- K fragments, then V fragments issued early (land during softmax)
    bf16x16 kf0[2], kf1[2], vb[4];
#pragma unroll
    for (int t = 0; t < 2; ++t) {
      const __bf16* kp = Kh + ((size_t)bh * SEQ + kv + t * 16 + ln) * DEPTH + half * 16;
      kf0[t] = *(const bf16x16*)kp;
      kf1[t] = *(const bf16x16*)(kp + 32);
    }
#pragma unroll
    for (int d = 0; d < 4; ++d) {
      const __bf16* vp = Vt + ((size_t)bh * DEPTH + d * 16 + ln) * SEQ + kv + half * 16;
      vb[d] = *(const bf16x16*)vp;
    }

    // ---- logits: Q (16x64) x K^T (64x32), 4 WMMAs
    f32x8 lc[2] = {};
#pragma unroll
    for (int t = 0; t < 2; ++t) {
      lc[t] = wmma_bf16(qa0, kf0[t], lc[t]);
      lc[t] = wmma_bf16(qa1, kf1[t], lc[t]);
    }

    // ---- base-2 scale + causal mask (C-layout: row = mr0+r, col = t*16+ln)
    float p[2][8];
#pragma unroll
    for (int t = 0; t < 2; ++t)
#pragma unroll
      for (int r = 0; r < 8; ++r) {
        float val = lc[t][r] * SCL2;
        if (causal) {
          int qi = qbase + mr0 + r;
          int kj = kv + t * 16 + ln;
          if (kj > qi) val = -1.0e10f;
        }
        p[t][r] = val;
      }

    // ---- online softmax row stats (reduce across the 16-lane half)
    float mnew[8], alpha[8];
#pragma unroll
    for (int r = 0; r < 8; ++r) {
      float tm = fmaxf(p[0][r], p[1][r]);
      tm = fmaxf(tm, __shfl_xor(tm, 1, 32));
      tm = fmaxf(tm, __shfl_xor(tm, 2, 32));
      tm = fmaxf(tm, __shfl_xor(tm, 4, 32));
      tm = fmaxf(tm, __shfl_xor(tm, 8, 32));
      mnew[r]  = fmaxf(mrow[r], tm);
      alpha[r] = __builtin_amdgcn_exp2f(mrow[r] - mnew[r]);
      mrow[r]  = mnew[r];
    }
#pragma unroll
    for (int t = 0; t < 2; ++t)
#pragma unroll
      for (int r = 0; r < 8; ++r)
        p[t][r] = __builtin_amdgcn_exp2f(p[t][r] - mnew[r]);
#pragma unroll
    for (int r = 0; r < 8; ++r) {
      float ts = p[0][r] + p[1][r];
      ts += __shfl_xor(ts, 1, 32);
      ts += __shfl_xor(ts, 2, 32);
      ts += __shfl_xor(ts, 4, 32);
      ts += __shfl_xor(ts, 8, 32);
      lrow[r] = lrow[r] * alpha[r] + ts;
    }
#pragma unroll
    for (int d = 0; d < 4; ++d)
#pragma unroll
      for (int r = 0; r < 8; ++r)
        o[d][r] *= alpha[r];

    // ---- P: C-layout -> LDS (row-major 16x32 bf16) -> A-fragment layout
    __bf16* pl = plds[wave];
#pragma unroll
    for (int t = 0; t < 2; ++t)
#pragma unroll
      for (int r = 0; r < 8; ++r)
        pl[(mr0 + r) * 32 + t * 16 + ln] = (__bf16)p[t][r];
    asm volatile("s_wait_dscnt 0x0" ::: "memory");
    bf16x16 pa;
    {
      const __bf16* pr = pl + ln * 32 + half * 16;
      bf16x8 c0 = *(const bf16x8*)pr;
      bf16x8 c1 = *(const bf16x8*)(pr + 8);
#pragma unroll
      for (int i = 0; i < 8; ++i) { pa[i] = c0[i]; pa[8 + i] = c1[i]; }
    }

    // ---- O += P (16x32) x V (32x64), 4 WMMAs (V fragments already resident)
#pragma unroll
    for (int d = 0; d < 4; ++d)
      o[d] = wmma_bf16(pa, vb[d], o[d]);
  }

  // ---- normalize (v_rcp + mul) + write merged-head bf16 [B,S,D]
  float rinv[8];
#pragma unroll
  for (int r = 0; r < 8; ++r) rinv[r] = __builtin_amdgcn_rcpf(lrow[r]);
  const int b_idx = bh >> 4, h = bh & 15;
#pragma unroll
  for (int d = 0; d < 4; ++d)
#pragma unroll
    for (int r = 0; r < 8; ++r) {
      float val = o[d][r] * rinv[r];
      int s = qbase + mr0 + r;
      size_t off = ((size_t)b_idx * SEQ + s) * DMODEL + h * DEPTH + d * 16 + ln;
      attn[off] = (__bf16)val;
    }
}

// ---------------------------------------------------------------------------
// Kernel 4: output projection  attn(bf16) @ WoT(bf16) + bo -> fp32 d_out.
// Same triple-buffered async A staging (bf16 tile, b32 per thread) + B
// register double-buffering.
// ---------------------------------------------------------------------------
__global__ __launch_bounds__(256) void oproj_kernel(const __bf16* __restrict__ X,
                                                    const __bf16* __restrict__ Wt,
                                                    const float* __restrict__ bias,
                                                    float* __restrict__ out) {
  __shared__ __bf16 Abuf[3][16 * 32];                // 3 x 1KB
  const int tid     = threadIdx.x;
  const int lane    = tid & 31;
  const int wave    = tid >> 5;
  const int rowbase = blockIdx.x * 16;
  const int colbase = blockIdx.y * 512 + wave * 64;
  const int half    = lane >> 4;
  const int ln      = lane & 15;

  // uniform staging: every thread copies 4B (2 bf16)
  const __bf16* gsrc = X + (size_t)(rowbase + (tid >> 4)) * DMODEL + (tid & 15) * 2;
  const uint32_t myl = (uint32_t)(uintptr_t)&Abuf[0][0] + (uint32_t)tid * 4u;

  async_b32(myl, gsrc);
  async_b32(myl + 1024u, gsrc + 32);

  const __bf16* wbase[4];
#pragma unroll
  for (int t = 0; t < 4; ++t)
    wbase[t] = Wt + (size_t)(colbase + t * 16 + ln) * DMODEL + half * 16;
  bf16x16 bfrag[2][4];
#pragma unroll
  for (int t = 0; t < 4; ++t) bfrag[0][t] = *(const bf16x16*)wbase[t];

  f32x8 acc[4] = {};
  const int NK = DMODEL / 32;
#pragma unroll 2
  for (int it = 0; it < NK; ++it) {
    if (it == NK - 1) wait_async0(); else wait_async1();
    __syncthreads();
    if (it + 2 < NK)
      async_b32(myl + (uint32_t)((it + 2) % 3) * 1024u, gsrc + (it + 2) * 32);
    if (it + 1 < NK) {
#pragma unroll
      for (int t = 0; t < 4; ++t)
        bfrag[(it + 1) & 1][t] = *(const bf16x16*)(wbase[t] + (it + 1) * 32);
    }
    const __bf16* ar = &Abuf[it % 3][ln * 32 + half * 16];
    bf16x8 c0 = *(const bf16x8*)ar;
    bf16x8 c1 = *(const bf16x8*)(ar + 8);
    bf16x16 a;
#pragma unroll
    for (int i = 0; i < 8; ++i) { a[i] = c0[i]; a[8 + i] = c1[i]; }
#pragma unroll
    for (int t = 0; t < 4; ++t)
      acc[t] = wmma_bf16(a, bfrag[it & 1][t], acc[t]);
  }

  const int mr0 = half * 8;
#pragma unroll
  for (int t = 0; t < 4; ++t) {
    int c = colbase + t * 16 + ln;
    float bvv = bias[c];
#pragma unroll
    for (int r = 0; r < 8; ++r)
      out[(size_t)(rowbase + mr0 + r) * DMODEL + c] = acc[t][r] + bvv;
  }
}

// ---------------------------------------------------------------------------
extern "C" void kernel_launch(void* const* d_in, const int* in_sizes, int n_in,
                              void* d_out, int out_size, void* d_ws, size_t ws_size,
                              hipStream_t stream) {
  (void)in_sizes; (void)n_in; (void)out_size; (void)ws_size;
  const float* q   = (const float*)d_in[0];
  const float* v   = (const float*)d_in[1];
  const float* k   = (const float*)d_in[2];
  const float* Wq  = (const float*)d_in[3];
  const float* bq  = (const float*)d_in[4];
  const float* Wk  = (const float*)d_in[5];
  const float* bk  = (const float*)d_in[6];
  const float* Wv  = (const float*)d_in[7];
  const float* bv  = (const float*)d_in[8];
  const float* Wo  = (const float*)d_in[9];
  const float* bo  = (const float*)d_in[10];
  const int* maskp = (const int*)d_in[11];

  __bf16* ws = (__bf16*)d_ws;
  const size_t DD  = (size_t)DMODEL * DMODEL;
  const size_t BSD = (size_t)BDIM * SEQ * DMODEL;
  __bf16* wqT  = ws;
  __bf16* wkT  = ws + DD;
  __bf16* wvT  = ws + 2 * DD;
  __bf16* woT  = ws + 3 * DD;
  __bf16* Qh   = ws + 4 * DD;                          // [B,H,S,64]
  __bf16* Kh   = Qh + BSD;                             // [B,H,S,64]
  __bf16* Vt   = Kh + BSD;                             // [B,H,64,S]
  __bf16* attn = Vt + BSD;                             // [B,S,D]

  dim3 tgrid(DMODEL / 32, DMODEL / 32);
  wt_convert_kernel<<<tgrid, 256, 0, stream>>>(Wq, wqT);
  wt_convert_kernel<<<tgrid, 256, 0, stream>>>(Wk, wkT);
  wt_convert_kernel<<<tgrid, 256, 0, stream>>>(Wv, wvT);
  wt_convert_kernel<<<tgrid, 256, 0, stream>>>(Wo, woT);

  dim3 pgrid(BDIM * SEQ / 16, DMODEL / 512);           // (512, 2)
  proj_kernel<<<pgrid, 256, 0, stream>>>(q, wqT, bq, Qh, 0);
  proj_kernel<<<pgrid, 256, 0, stream>>>(k, wkT, bk, Kh, 0);
  proj_kernel<<<pgrid, 256, 0, stream>>>(v, wvT, bv, Vt, 1);

  flash_kernel<<<dim3(BDIM * NHEAD * (SEQ / 16) / 8), 256, 0, stream>>>(Qh, Kh, Vt, attn, maskp);

  oproj_kernel<<<pgrid, 256, 0, stream>>>(attn, woT, bo, (float*)d_out);
}
